// dense_block_8555574854009
// MI455X (gfx1250) — matrix operand order repaired and checked by reading, not compiled, not run on hardware
//
#include <hip/hip_runtime.h>

#define NBATCH 8
#define HWSZ   3136        // 56*56
#define IMGW   56
#define PTOT   25088       // NBATCH*HWSZ
#define C0     256
#define GROWTH 32
#define CMID   128
#define COUT   448
#define NLAYER 6
#define EPSV   1.1e-05f
#define LDSROW 40          // ushorts per LDS row (80 bytes, bank-conflict-free, 16B aligned)
#define TILEN  64

typedef __bf16 bf16_t;
typedef __attribute__((ext_vector_type(16))) bf16_t v16bf;
typedef __attribute__((ext_vector_type(8)))  bf16_t v8bf;
typedef __attribute__((ext_vector_type(8)))  float  v8f;

__device__ __forceinline__ unsigned short f2bf(float f) {
  unsigned u = __builtin_bit_cast(unsigned, f);
  unsigned r = u + 0x7FFFu + ((u >> 16) & 1u);   // round-to-nearest-even
  return (unsigned short)(r >> 16);
}

// Build a v16bf B-fragment from two aligned 16B LDS chunks.
__device__ __forceinline__ v16bf load_b_frag(const unsigned short* bt, int row, int half) {
  const v8bf* p = (const v8bf*)(bt + row * LDSROW + half * 8);
  v8bf lo = p[0];
  v8bf hi = p[2];
  v16bf b;
#pragma unroll
  for (int i = 0; i < 8; ++i) { b[i] = lo[i]; b[i + 8] = hi[i]; }
  return b;
}

// ---------------- input copy: x -> d_out channels [0,256) ----------------
__global__ void copy_in_kernel(const float* __restrict__ x, float* __restrict__ out) {
  int i = blockIdx.x * 256 + threadIdx.x;          // exactly NBATCH*C0*HWSZ threads
  int nb  = i / (C0 * HWSZ);
  int rem = i - nb * (C0 * HWSZ);
  out[(size_t)nb * COUT * HWSZ + rem] = x[i];
}

// ------------- BN batch-stats + fused affine (scale/shift) ---------------
// One block per channel. Handles both feats layout (cstride=HWSZ, bstride=COUT*HWSZ,
// nbat=8, cnt=HWSZ) and mid layout (cstride=PTOT, bstride=0, nbat=1, cnt=PTOT).
__global__ void __launch_bounds__(256) stats_kernel(
    const float* __restrict__ src, int cstride, int bstride, int nbat, int cnt,
    const float* __restrict__ bn_g, const float* __restrict__ bn_b,
    const float* __restrict__ s_g,  const float* __restrict__ s_b,
    float* __restrict__ sc, float* __restrict__ sh) {
  int c = blockIdx.x;
  const float* base = src + (long long)c * cstride;
  float s = 0.f, q = 0.f;
  for (int b = 0; b < nbat; ++b) {
    const float* p = base + (long long)b * bstride;
    for (int i = threadIdx.x; i < cnt; i += 256) { float v = p[i]; s += v; q += v * v; }
  }
  __shared__ float ls[256], lq[256];
  ls[threadIdx.x] = s; lq[threadIdx.x] = q;
  __syncthreads();
  for (int off = 128; off > 0; off >>= 1) {
    if (threadIdx.x < off) { ls[threadIdx.x] += ls[threadIdx.x + off]; lq[threadIdx.x] += lq[threadIdx.x + off]; }
    __syncthreads();
  }
  if (threadIdx.x == 0) {
    float nTot = (float)nbat * (float)cnt;
    float mean = ls[0] / nTot;
    float var  = lq[0] / nTot - mean * mean;      // biased variance (jnp.var)
    float inv  = rsqrtf(var + EPSV);
    sc[c] = bn_g[c] * inv * s_g[c];
    sh[c] = (bn_b[c] - mean * bn_g[c] * inv) * s_g[c] + s_b[c];
  }
}

// ------- repack w1 (CMID x cin) into WMMA A-fragment order, bf16 ----------
// frag index = ((mt*nk + kc)*32 + lane)*16 + e ; k(e,half) per ISA 16-bit A 16x32 layout
__global__ void conv_w1_kernel(const float* __restrict__ w1, unsigned short* __restrict__ w1f, int cin) {
  int nk = cin >> 5;
  int tid = blockIdx.x * 256 + threadIdx.x;
  int total = 8 * nk * 512;
  if (tid >= total) return;
  int e    = tid & 15;
  int lane = (tid >> 4) & 31;
  int kc   = (tid >> 9) % nk;
  int mt   = (tid >> 9) / nk;
  int m = lane & 15, half = lane >> 4;
  int k = (e < 8 ? e : e + 8) + half * 8;
  w1f[tid] = f2bf(w1[(size_t)(mt * 16 + m) * cin + kc * 32 + k]);
}

// ---- repack w2 (32 x 128 x 3 x 3) into A-fragment order with k = s*128+c ----
__global__ void conv_w2_kernel(const float* __restrict__ w2, unsigned short* __restrict__ w2f) {
  int tid = blockIdx.x * 256 + threadIdx.x;       // total 2*36*512 = 36864
  if (tid >= 2 * 36 * 512) return;
  int e    = tid & 15;
  int lane = (tid >> 4) & 31;
  int kc   = (tid >> 9) % 36;
  int mt   = (tid >> 9) / 36;
  int m = lane & 15, half = lane >> 4;
  int k  = (e < 8 ? e : e + 8) + half * 8;
  int kg = kc * 32 + k;
  int s  = kg >> 7;            // spatial tap 0..8
  int c  = kg & 127;           // input channel
  w2f[tid] = f2bf(w2[(size_t)(mt * 16 + m) * 1152 + c * 9 + s]);
}

// ------------------- GEMM1: 1x1 conv fused with BN1+ReLU -------------------
// out mid[co, p] = sum_ci w1[co,ci] * relu(feats[ci,p]*sc[ci]+sh[ci])
// block: M=128 x N=64 ; 8 waves, wave = one 16-row m-tile, 4 n-subtiles
__global__ void __launch_bounds__(256) gemm1_kernel(
    const float* __restrict__ feats, const unsigned short* __restrict__ w1f,
    const float* __restrict__ sc, const float* __restrict__ sh,
    float* __restrict__ mid, int cin) {
  __shared__ unsigned short Bt[TILEN * LDSROW];
  int nk  = cin >> 5;
  int p0  = blockIdx.x * TILEN;
  int nb  = p0 / HWSZ;
  int hw0 = p0 - nb * HWSZ;                       // tile never crosses image (3136 % 64 == 0)
  const float* fb = feats + (size_t)nb * COUT * HWSZ + hw0;

  int t = threadIdx.x, lane = t & 31, wid = t >> 5;
  int mrow = lane & 15, half = lane >> 4;
  v8f acc[4] = {};
  const unsigned short* abase = w1f + ((size_t)wid * nk) * 512 + lane * 16;

  for (int kc = 0; kc < nk; ++kc) {
    __syncthreads();
    { // cooperative stage: 2048 elems, 8 per thread; fused affine+ReLU+bf16 cvt
      int n = t & 63, kq = t >> 6;
      const float* col = fb + n;
#pragma unroll
      for (int j = 0; j < 8; ++j) {
        int kl = kq * 8 + j;
        int ci = kc * 32 + kl;
        float v = col[(size_t)ci * HWSZ];
        v = v * sc[ci] + sh[ci];
        v = v > 0.f ? v : 0.f;
        Bt[n * LDSROW + kl] = f2bf(v);
      }
    }
    __syncthreads();
    v16bf a = *(const v16bf*)(abase + (size_t)kc * 512);
#pragma unroll
    for (int ns = 0; ns < 4; ++ns) {
      v16bf b = load_b_frag(Bt, ns * 16 + mrow, half);
      acc[ns] = __builtin_amdgcn_wmma_f32_16x16x32_bf16(
          false, a, false, b, (short)0, acc[ns], false, false);
    }
  }
  // C/D layout: VGPR r holds M=r (lanes 0-15) / M=r+8 (lanes 16-31), N=lane&15
#pragma unroll
  for (int ns = 0; ns < 4; ++ns)
#pragma unroll
    for (int r = 0; r < 8; ++r) {
      int M = r + half * 8;
      int N = ns * 16 + mrow;
      mid[(size_t)(wid * 16 + M) * PTOT + p0 + N] = acc[ns][r];
    }
}

// --------- GEMM2: implicit 3x3 conv fused with BN2+ReLU + zero pad ---------
// out[g,p] = sum_{s,c} w2[g,c,s] * act2[c, p+tap(s)] ; M=32, K=1152, N=25088
// block: M=32 x N=64 ; 8 waves = 2 m-tiles x 4 n-subtiles, 1 acc each
__global__ void __launch_bounds__(256) gemm2_kernel(
    const float* __restrict__ mid, const unsigned short* __restrict__ w2f,
    const float* __restrict__ sc, const float* __restrict__ sh,
    float* __restrict__ out, int cbase) {
  __shared__ unsigned short Bt[TILEN * LDSROW];
  int p0  = blockIdx.x * TILEN;
  int nb  = p0 / HWSZ;
  int hw0 = p0 - nb * HWSZ;
  const float* midb = mid + (size_t)nb * HWSZ;

  int t = threadIdx.x, lane = t & 31, wid = t >> 5;
  int mt = wid >> 2, ns = wid & 3;
  int mrow = lane & 15, half = lane >> 4;
  v8f acc = {};
  const unsigned short* abase = w2f + ((size_t)mt * 36) * 512 + lane * 16;

  for (int kc = 0; kc < 36; ++kc) {
    __syncthreads();
    { // stage with im2col gather: fixed spatial tap per K=32 chunk
      int n = t & 63, kq = t >> 6;
      int s  = kc >> 2;
      int dy = s / 3 - 1, dx = s % 3 - 1;
      int hw = hw0 + n;
      int h = hw / IMGW + dy;
      int w = hw % IMGW + dx;
      bool ok = ((unsigned)h < (unsigned)IMGW) && ((unsigned)w < (unsigned)IMGW);
      int hw2 = h * IMGW + w;
#pragma unroll
      for (int j = 0; j < 8; ++j) {
        int kl = kq * 8 + j;
        int c  = ((kc & 3) << 5) + kl;
        float v = 0.f;
        if (ok) {                       // pad with zeros AFTER affine+ReLU (conv input)
          float m_ = midb[(size_t)c * PTOT + hw2];
          v = m_ * sc[c] + sh[c];
          v = v > 0.f ? v : 0.f;
        }
        Bt[n * LDSROW + kl] = f2bf(v);
      }
    }
    __syncthreads();
    v16bf a = *(const v16bf*)(abase + (size_t)kc * 512);
    v16bf b = load_b_frag(Bt, ns * 16 + mrow, half);
    acc = __builtin_amdgcn_wmma_f32_16x16x32_bf16(
        false, a, false, b, (short)0, acc, false, false);
  }
  float* ob = out + (size_t)nb * COUT * HWSZ + hw0;
#pragma unroll
  for (int r = 0; r < 8; ++r) {
    int M = r + half * 8;
    int N = ns * 16 + mrow;
    ob[(size_t)(cbase + mt * 16 + M) * HWSZ + N] = acc[r];
  }
}

extern "C" void kernel_launch(void* const* d_in, const int* in_sizes, int n_in,
                              void* d_out, int out_size, void* d_ws, size_t ws_size,
                              hipStream_t stream) {
  (void)in_sizes; (void)n_in; (void)out_size; (void)ws_size;
  const float* x = (const float*)d_in[0];
  float* out = (float*)d_out;

  // workspace carve-up (all 256B aligned): mid, w1frag, w2frag, sc/sh arrays
  char* ws = (char*)d_ws;
  size_t off = 0;
  float* mid = (float*)(ws + off);            off += ((size_t)CMID * PTOT * 4 + 255) & ~(size_t)255;
  unsigned short* w1f = (unsigned short*)(ws + off); off += ((size_t)8 * 13 * 512 * 2 + 255) & ~(size_t)255;
  unsigned short* w2f = (unsigned short*)(ws + off); off += ((size_t)2 * 36 * 512 * 2 + 255) & ~(size_t)255;
  float* sc1 = (float*)(ws + off); off += 2048;
  float* sh1 = (float*)(ws + off); off += 2048;
  float* sc2 = (float*)(ws + off); off += 2048;
  float* sh2 = (float*)(ws + off); off += 2048;

  // seed feats (channels [0,256)) directly into d_out — concat is free
  copy_in_kernel<<<dim3((NBATCH * C0 * HWSZ) / 256), dim3(256), 0, stream>>>(x, out);

  for (int i = 0; i < NLAYER; ++i) {
    int cin = C0 + i * GROWTH;
    // params flattened: x, then per layer {bn1_g,bn1_b,s1_g,s1_b,w1,bn2_g,bn2_b,s2_g,s2_b,w2}
    const float* prm[10];
    for (int j = 0; j < 10; ++j) prm[j] = (const float*)d_in[1 + i * 10 + j];

    stats_kernel<<<dim3(cin), dim3(256), 0, stream>>>(
        out, HWSZ, COUT * HWSZ, NBATCH, HWSZ, prm[0], prm[1], prm[2], prm[3], sc1, sh1);
    conv_w1_kernel<<<dim3((8 * (cin / 32) * 512 + 255) / 256), dim3(256), 0, stream>>>(
        prm[4], w1f, cin);
    gemm1_kernel<<<dim3(PTOT / TILEN), dim3(256), 0, stream>>>(
        out, w1f, sc1, sh1, mid, cin);
    stats_kernel<<<dim3(CMID), dim3(256), 0, stream>>>(
        mid, PTOT, 0, 1, PTOT, prm[5], prm[6], prm[7], prm[8], sc2, sh2);
    conv_w2_kernel<<<dim3((2 * 36 * 512 + 255) / 256), dim3(256), 0, stream>>>(prm[9], w2f);
    gemm2_kernel<<<dim3(PTOT / TILEN), dim3(256), 0, stream>>>(
        mid, w2f, sc2, sh2, out, cin);
  }
}
